// IthemalRNN_88905823027395
// MI455X (gfx1250) — matrix operand order repaired and checked by reading, not compiled, not run on hardware
//
#include <hip/hip_runtime.h>
#include <hip/hip_bf16.h>
#include <stdint.h>

// Problem sizes (fixed by reference)
#define TT 16
#define NN 4096
#define EE 256
#define HH 256
#define G4 1024           // 4*H
#define KK (EE + HH)      // 512

typedef __attribute__((ext_vector_type(16))) __bf16          v16bf;
typedef __attribute__((ext_vector_type(8)))  float           v8f;
typedef __attribute__((ext_vector_type(8)))  unsigned short  us8;
typedef unsigned short u16;

union V16 { v16bf v; u16 s[16]; us8 h[2]; };

__device__ __forceinline__ u16 f2bf(float f) {
  unsigned u = __builtin_bit_cast(unsigned, f);
  u += 0x7FFFu + ((u >> 16) & 1u);     // round-to-nearest-even
  return (u16)(u >> 16);
}

__device__ __forceinline__ float sigf(float x) { return 1.0f / (1.0f + __expf(-x)); }

// ---------------------------------------------------------------------------
// Prep kernels
// ---------------------------------------------------------------------------

// Xbf[t][n][e] = bf16(emb[token_ids[n][t]][e])   (T*N*E = 2^24 elements)
__global__ void k_gather(const int* __restrict__ tok, const float* __restrict__ emb,
                         u16* __restrict__ Xbf) {
  int idx = blockIdx.x * 256 + threadIdx.x;
  int e    = idx & (EE - 1);
  int nIdx = (idx >> 8) & (NN - 1);
  int t    = idx >> 20;
  int token = tok[nIdx * TT + t];
  Xbf[idx] = f2bf(emb[(size_t)token * EE + e]);
}

// Wcat[n][k] = bf16( k<E ? W_ih[n][k] : W_hh[n][k-E] )   (1024 x 512)
__global__ void k_wcat(const float* __restrict__ Wih, const float* __restrict__ Whh,
                       u16* __restrict__ Wcat) {
  int idx = blockIdx.x * 256 + threadIdx.x;     // 0 .. 1024*512-1
  int n = idx >> 9;
  int k = idx & (KK - 1);
  float v = (k < EE) ? Wih[(size_t)n * EE + k] : Whh[(size_t)n * HH + (k - EE)];
  Wcat[idx] = f2bf(v);
}

// bf16 copy of ins_W_ih (1024 x 256)
__global__ void k_w16(const float* __restrict__ W, u16* __restrict__ Wb) {
  int idx = blockIdx.x * 256 + threadIdx.x;
  Wb[idx] = f2bf(W[idx]);
}

__global__ void k_zero(unsigned* __restrict__ p, int n) {
  int i = blockIdx.x * 256 + threadIdx.x;
  if (i < n) p[i] = 0u;
}

// ---------------------------------------------------------------------------
// Token LSTM: one launch per timestep.
// grid = (N/16, H/16), block = 128 (4 waves; wave g computes gate g's 16x16 tile)
// gates = [x_t | h_{t-1}] @ Wcat^T + b ; then fused LSTM pointwise update.
// ---------------------------------------------------------------------------
__global__ void k_tok_step(int t,
                           const u16* __restrict__ Xbf,   // [T][N][E] bf16
                           const u16* __restrict__ Hin,   // [N][H]    bf16
                           u16* __restrict__ Hout,        // [N][H]    bf16
                           float* __restrict__ Cbuf,      // [N][H]    f32
                           const u16* __restrict__ Wcat,  // [4H][K]   bf16
                           const float* __restrict__ tok_b,
                           const int* __restrict__ lengths,
                           u16* __restrict__ Feat)        // [N][H]    bf16
{
  __shared__ float glds[4][16][16];

  const int wave = threadIdx.x >> 5;
  const int lane = threadIdx.x & 31;
  const int m0  = blockIdx.x * 16;
  const int nh0 = blockIdx.y * 16;

  const int mrow = m0 + (lane & 15);                 // A-matrix row for this lane
  const int ncol = wave * HH + nh0 + (lane & 15);    // output/gate column
  const u16* brow = Wcat + (size_t)ncol * KK;

  const int aoff  = (lane & 16) ? 8  : 0;            // A lane K-sub-offset
  const int boff  = (lane & 16) ? 8  : 0;            // B lane K-sub-offset
  const int mbase = (lane & 16) ? 8  : 0;            // C/D row base

  v8f acc = {};
  for (int kb = 0; kb < KK; kb += 32) {
    const u16* arow = (kb < EE)
        ? (Xbf + ((size_t)t * NN + mrow) * EE + kb)
        : (Hin + (size_t)mrow * HH + (kb - EE));
    V16 a, b;
    a.h[0] = *(const us8*)(arow + aoff);
    a.h[1] = *(const us8*)(arow + aoff + 16);
    b.h[0] = *(const us8*)(brow + kb + boff);
    b.h[1] = *(const us8*)(brow + kb + boff + 16);
    acc = __builtin_amdgcn_wmma_f32_16x16x32_bf16(false, a.v, false, b.v,
                                                  (short)0, acc, false, false);
  }

  const float bias = tok_b[ncol];
#pragma unroll
  for (int r = 0; r < 8; ++r)
    glds[wave][mbase + r][lane & 15] = acc[r] + bias;

  __syncthreads();

  // LSTM pointwise: 256 (m,j) cells, 128 threads -> 2 each
  for (int idx = threadIdx.x; idx < 256; idx += 128) {
    int ml = idx >> 4, nl = idx & 15;
    float ig = sigf (glds[0][ml][nl]);
    float fg = sigf (glds[1][ml][nl]);
    float gg = tanhf(glds[2][ml][nl]);
    float og = sigf (glds[3][ml][nl]);
    int m = m0 + ml, j = nh0 + nl;
    size_t o = (size_t)m * HH + j;
    float c = fg * Cbuf[o] + ig * gg;
    float h = og * tanhf(c);
    Cbuf[o] = c;
    Hout[o] = f2bf(h);
    if (lengths[m] - 1 == t) Feat[o] = f2bf(h);
  }
}

// ---------------------------------------------------------------------------
// PreIns = Feat @ ins_W_ih^T + ins_b   (4096 x 1024, K=256)
// grid = (N/16, 4H/16), block = 32 (one wave per 16x16 tile)
// ---------------------------------------------------------------------------
__global__ void k_ins_pre(const u16* __restrict__ Feat,   // [N][H]  bf16
                          const u16* __restrict__ Wins,   // [4H][H] bf16
                          const float* __restrict__ ins_b,
                          float* __restrict__ Pre)        // [N][4H] f32
{
  const int lane = threadIdx.x & 31;
  const int m0 = blockIdx.x * 16;
  const int n0 = blockIdx.y * 16;

  const int mrow = m0 + (lane & 15);
  const int ncol = n0 + (lane & 15);
  const u16* arow = Feat + (size_t)mrow * HH;
  const u16* brow = Wins + (size_t)ncol * HH;

  const int aoff  = (lane & 16) ? 8  : 0;
  const int boff  = (lane & 16) ? 8  : 0;
  const int mbase = (lane & 16) ? 8  : 0;

  v8f acc = {};
  for (int kb = 0; kb < HH; kb += 32) {
    V16 a, b;
    a.h[0] = *(const us8*)(arow + kb + aoff);
    a.h[1] = *(const us8*)(arow + kb + aoff + 16);
    b.h[0] = *(const us8*)(brow + kb + boff);
    b.h[1] = *(const us8*)(brow + kb + boff + 16);
    acc = __builtin_amdgcn_wmma_f32_16x16x32_bf16(false, a.v, false, b.v,
                                                  (short)0, acc, false, false);
  }

  const float bias = ins_b[ncol];
#pragma unroll
  for (int r = 0; r < 8; ++r)
    Pre[(size_t)(m0 + mbase + r) * G4 + ncol] = acc[r] + bias;
}

// ---------------------------------------------------------------------------
// Instruction LSTM: strictly sequential 4096 steps, batch 1. One block, 1024 thr.
// Thread n computes gate n = PreIns[t][n] + h . ins_W_hh[n]; h,c live in LDS.
// Ends with h . lin_W + lin_b -> scalar output.
// ---------------------------------------------------------------------------
__global__ void k_ins_seq(const float* __restrict__ Pre,   // [N][4H]
                          const float* __restrict__ Whh,   // [4H][H] f32
                          const float* __restrict__ linW,  // [H]
                          const float* __restrict__ linb,  // [1]
                          float* __restrict__ out)
{
  __shared__ float hl[HH];
  __shared__ float cl[HH];
  __shared__ float gl[G4];

  const int tid = threadIdx.x;
  if (tid < HH) { hl[tid] = 0.0f; cl[tid] = 0.0f; }
  __syncthreads();

  const float* wrow = Whh + (size_t)tid * HH;

  for (int t = 0; t < NN; ++t) {
    float s = Pre[(size_t)t * G4 + tid];
#pragma unroll 8
    for (int k = 0; k < HH; ++k) s += hl[k] * wrow[k];
    gl[tid] = s;
    __syncthreads();

    if (tid < HH) {
      float ig = sigf (gl[tid]);
      float fg = sigf (gl[HH + tid]);
      float gg = tanhf(gl[2 * HH + tid]);
      float og = sigf (gl[3 * HH + tid]);
      float c  = fg * cl[tid] + ig * gg;
      cl[tid] = c;
      hl[tid] = og * tanhf(c);
    }
    __syncthreads();
  }

  if (tid < HH) gl[tid] = hl[tid] * linW[tid];
  __syncthreads();
  for (int off = HH / 2; off > 0; off >>= 1) {
    if (tid < off) gl[tid] += gl[tid + off];
    __syncthreads();
  }
  if (tid == 0) out[0] = gl[0] + linb[0];
}

// ---------------------------------------------------------------------------
// Host launcher
// ---------------------------------------------------------------------------
extern "C" void kernel_launch(void* const* d_in, const int* in_sizes, int n_in,
                              void* d_out, int out_size, void* d_ws, size_t ws_size,
                              hipStream_t stream) {
  (void)in_sizes; (void)n_in; (void)out_size; (void)ws_size;

  const int*   token_ids = (const int*)  d_in[0];   // [N,T]
  const int*   lengths   = (const int*)  d_in[1];   // [N]
  const float* emb       = (const float*)d_in[2];   // [V,E]
  const float* tok_W_ih  = (const float*)d_in[3];   // [4H,E]
  const float* tok_W_hh  = (const float*)d_in[4];   // [4H,H]
  const float* tok_b     = (const float*)d_in[5];   // [4H]
  const float* ins_W_ih  = (const float*)d_in[6];   // [4H,H]
  const float* ins_W_hh  = (const float*)d_in[7];   // [4H,H]
  const float* ins_b     = (const float*)d_in[8];   // [4H]
  const float* lin_W     = (const float*)d_in[9];   // [1,H]
  const float* lin_b     = (const float*)d_in[10];  // [1]
  float* out = (float*)d_out;

  // Workspace layout (bytes; total = 62,390,272)
  char* ws = (char*)d_ws;
  u16*   Xbf  = (u16*)  (ws + 0);          // T*N*E bf16 = 33,554,432
  u16*   Wcat = (u16*)  (ws + 33554432);   // 4H*K bf16  =  1,048,576
  u16*   Wins = (u16*)  (ws + 34603008);   // 4H*H bf16  =    524,288
  u16*   H0   = (u16*)  (ws + 35127296);   // N*H bf16   =  2,097,152
  u16*   H1   = (u16*)  (ws + 37224448);   // N*H bf16   =  2,097,152
  float* Cb   = (float*)(ws + 39321600);   // N*H f32    =  4,194,304
  u16*   Feat = (u16*)  (ws + 43515904);   // N*H bf16   =  2,097,152
  float* Pre  = (float*)(ws + 45613056);   // N*4H f32   = 16,777,216

  // Prep: gather/convert, pack weights, zero initial h/c
  k_gather<<<(TT * NN * EE) / 256, 256, 0, stream>>>(token_ids, emb, Xbf);
  k_wcat  <<<(G4 * KK) / 256,      256, 0, stream>>>(tok_W_ih, tok_W_hh, Wcat);
  k_w16   <<<(G4 * HH) / 256,      256, 0, stream>>>(ins_W_ih, Wins);
  k_zero  <<<(NN * HH / 2) / 256,  256, 0, stream>>>((unsigned*)H0, NN * HH / 2);
  k_zero  <<<(NN * HH) / 256,      256, 0, stream>>>((unsigned*)Cb, NN * HH);

  // Token LSTM: 16 dependent steps (kernel boundary = device-wide barrier)
  dim3 gridTok(NN / 16, HH / 16);
  for (int t = 0; t < TT; ++t) {
    const u16* Hi = (t & 1) ? H1 : H0;
    u16*       Ho = (t & 1) ? H0 : H1;
    k_tok_step<<<gridTok, 128, 0, stream>>>(t, Xbf, Hi, Ho, Cb, Wcat,
                                            tok_b, lengths, Feat);
  }

  // Instruction LSTM: input-gate GEMM (parallel), then sequential recurrence
  dim3 gridPre(NN / 16, G4 / 16);
  k_ins_pre<<<gridPre, 32, 0, stream>>>(Feat, Wins, ins_b, Pre);
  k_ins_seq<<<1, 1024, 0, stream>>>(Pre, ins_W_hh, lin_W, lin_b, out);
}
